// SimpleSLSTMBlock_61005715473187
// MI455X (gfx1250) — matrix operand order repaired and checked by reading, not compiled, not run on hardware
//
#include <hip/hip_runtime.h>
#include <hip/hip_bf16.h>
#include <math.h>

// ---------------- problem constants ----------------
#define BB   4096
#define DD   2048
#define HH   2048
#define NBK  4
#define BIN  512
#define BOUT 512
#define EPSF 1e-5f

// ---------------- GEMM tiling ----------------
#define KB 32     // K step (matches wmma_f32_16x16x32_bf16)
#define MT 64     // workgroup M tile
#define NT 64     // workgroup N tile
#define NSTEP 32  // 2 phases x 16 K-steps

typedef __bf16 bf16;
typedef bf16  v8bf  __attribute__((ext_vector_type(8)));
typedef bf16  v16bf __attribute__((ext_vector_type(16)));
typedef float v8f   __attribute__((ext_vector_type(8)));

#define S_ACT   ((size_t)BB * DD)           // one bf16 activation plane
#define W_MAT   ((size_t)NBK * BIN * BOUT)  // one transposed weight matrix (elems)
#define W_HALF  ((size_t)4 * W_MAT)         // W-group or R-group (elems)

__device__ __forceinline__ v16bf cat16(v8bf lo, v8bf hi) {
  return __builtin_shufflevector(lo, hi, 0,1,2,3,4,5,6,7,8,9,10,11,12,13,14,15);
}
__device__ __forceinline__ float sigmoidf_(float v) { return 1.0f / (1.0f + expf(-v)); }

// CDNA5 async global->LDS copy (16B per lane), tracked by ASYNCcnt.
__device__ __forceinline__ void async_copy16(unsigned lds_off, const bf16* gptr) {
  asm volatile("global_load_async_to_lds_b128 %0, %1, off"
               :: "v"(lds_off), "v"(gptr) : "memory");
}
__device__ __forceinline__ void wait_async0() {
  asm volatile("s_wait_asynccnt 0" ::: "memory");
}

// =====================================================================
// Kernel 1: LayerNorm + causal conv3 + SiLU; cast x, x_conv, h_prev -> bf16
// =====================================================================
__global__ __launch_bounds__(256) void prep_kernel(
    const float* __restrict__ x, const float* __restrict__ h_prev,
    const float* __restrict__ ln_g, const float* __restrict__ ln_b,
    const float* __restrict__ conv_w, const float* __restrict__ conv_b,
    bf16* __restrict__ act)   // act: [xb | xcb | hb], each BB*DD
{
  __shared__ float xn[DD];
  __shared__ float red[256 * 2];

  const int tid = threadIdx.x;
  const int row = blockIdx.x;
  const float* xr = x + (size_t)row * DD;

  float s = 0.f, s2 = 0.f;
  for (int d = tid; d < DD; d += 256) { float v = xr[d]; s += v; s2 += v * v; }
  red[tid] = s; red[256 + tid] = s2;
  __syncthreads();
  for (int off = 128; off > 0; off >>= 1) {
    if (tid < off) { red[tid] += red[tid + off]; red[256 + tid] += red[256 + tid + off]; }
    __syncthreads();
  }
  const float mu   = red[0] * (1.0f / DD);
  const float var  = red[256] * (1.0f / DD) - mu * mu;
  const float rstd = rsqrtf(var + EPSF);

  bf16* xb  = act;
  bf16* xcb = act + S_ACT;
  bf16* hb  = act + 2 * S_ACT;

  for (int d = tid; d < DD; d += 256) {
    float v = xr[d];
    xn[d] = (v - mu) * rstd * ln_g[d] + ln_b[d];
    xb[(size_t)row * DD + d] = (bf16)v;
    hb[(size_t)row * DD + d] = (bf16)h_prev[(size_t)row * DD + d];
  }
  __syncthreads();

  const float w0 = conv_w[0], w1 = conv_w[1], w2 = conv_w[2], cb = conv_b[0];
  for (int d = tid; d < DD; d += 256) {
    float a = (d >= 2) ? xn[d - 2] : 0.f;
    float b = (d >= 1) ? xn[d - 1] : 0.f;
    float c = xn[d];
    float xc = w0 * a + w1 * b + w2 * c + cb;
    xcb[(size_t)row * DD + d] = (bf16)(xc * sigmoidf_(xc));
  }
}

// =====================================================================
// Kernel 2: one-time weight transpose+convert.
// dst layout: [which(W=0,R=1)][gate][blk][n(512)][k(512)] bf16 (N-major)
// Grid: (16 jtiles, 16 ktiles, 32 = mat*4+blk), 256 threads.
// =====================================================================
__global__ __launch_bounds__(256) void wcvt_kernel(
    const float* __restrict__ Wz, const float* __restrict__ Wi,
    const float* __restrict__ Wf, const float* __restrict__ Wo,
    const float* __restrict__ Rz, const float* __restrict__ Ri,
    const float* __restrict__ Rf, const float* __restrict__ Ro,
    bf16* __restrict__ dst)
{
  __shared__ float tile[32][33];
  const float* srcs[8] = {Wz, Wi, Wf, Wo, Rz, Ri, Rf, Ro};

  const int mat = blockIdx.z >> 2;           // 0..7
  const int blk = blockIdx.z & 3;
  const int k0  = blockIdx.y * 32;
  const int j0  = blockIdx.x * 32;
  const int tx  = threadIdx.x & 31;
  const int ty  = threadIdx.x >> 5;          // 0..7

  const float* src = srcs[mat] + (size_t)blk * BIN * BOUT;
  #pragma unroll
  for (int i = 0; i < 4; ++i)
    tile[ty + i * 8][tx] = src[(size_t)(k0 + ty + i * 8) * BOUT + j0 + tx];
  __syncthreads();

  const size_t dbase = (size_t)(mat >> 2) * W_HALF + (size_t)(mat & 3) * W_MAT
                     + (size_t)blk * BIN * BOUT;
  #pragma unroll
  for (int i = 0; i < 4; ++i) {
    int j = j0 + ty + i * 8;
    dst[dbase + (size_t)j * BIN + k0 + tx] = (bf16)tile[tx][ty + i * 8];
  }
}

// =====================================================================
// Kernel 3: fused 8-GEMM (4 gates x {W,R}) + sLSTM epilogue.
// Double-buffered async global->LDS pipeline.
// Grid: (H/NT, B/MT) = (32, 64), 256 threads = 8 wave32 waves (2x4).
// =====================================================================
__global__ __launch_bounds__(256) void gemm_gates_kernel(
    const bf16* __restrict__ act,   // [xb | xcb | hb]
    const bf16* __restrict__ wt,    // transposed bf16 weights
    const float* __restrict__ bz_, const float* __restrict__ bi_,
    const float* __restrict__ bf_, const float* __restrict__ bo_,
    const float* __restrict__ rbz_, const float* __restrict__ rbi_,
    const float* __restrict__ rbf_, const float* __restrict__ rbo_,
    const float* __restrict__ c_prev,
    float* __restrict__ h_out, float* __restrict__ c_out)
{
  __shared__ bf16 lds_a1[2][MT * KB];        // x (ph0) / h (ph1)      2x4 KB
  __shared__ bf16 lds_a2[2][MT * KB];        // x_conv (ph0 only)      2x4 KB
  __shared__ bf16 lds_b[2][4][NT * KB];      // per-gate B, N-major   2x16 KB

  const int tid  = threadIdx.x;
  const int lane = tid & 31;
  const int wave = tid >> 5;
  const int wm   = wave & 1;
  const int wn   = wave >> 1;

  const int m_base   = blockIdx.y * MT;
  const int n_global = blockIdx.x * NT;
  const int blk      = n_global >> 9;
  const int n_in_blk = n_global & (BOUT - 1);

  // per-thread staging coordinates (16B granules)
  const int sr  = tid >> 2;            // 0..63 : A row / B n-row
  const int sj  = (tid & 3) * 8;       // element offset within 32-wide K slice
  const unsigned a1b = (unsigned)(size_t)&lds_a1[0][0];
  const unsigned a2b = (unsigned)(size_t)&lds_a2[0][0];
  const unsigned bb  = (unsigned)(size_t)&lds_b[0][0][0];

  // stage step s into buffer buf: 5-6 async 16B copies per thread
  auto stage = [&](int s, int buf) {
    const int ph = s >> 4;
    const int k0 = (s & 15) * KB;
    const size_t act1 = (size_t)(ph ? 2 : 0) * S_ACT;   // x or h
    const size_t arow = (size_t)(m_base + sr) * DD + blk * BIN + k0 + sj;
    async_copy16(a1b + (unsigned)(buf * MT * KB + sr * KB + sj) * 2,
                 act + act1 + arow);
    if (!ph)
      async_copy16(a2b + (unsigned)(buf * MT * KB + sr * KB + sj) * 2,
                   act + S_ACT + arow);
    const size_t wrow = (size_t)ph * W_HALF + (size_t)blk * BIN * BOUT
                      + (size_t)(n_in_blk + sr) * BIN + k0 + sj;
    #pragma unroll
    for (int g = 0; g < 4; ++g)
      async_copy16(bb + (unsigned)(((buf * 4 + g) * NT + sr) * KB + sj) * 2,
                   wt + wrow + (size_t)g * W_MAT);
  };

  v8f acc[4][2];
  #pragma unroll
  for (int g = 0; g < 4; ++g)
    #pragma unroll
    for (int mt = 0; mt < 2; ++mt) acc[g][mt] = (v8f){};

  stage(0, 0);

  #pragma unroll 1
  for (int s = 0; s < NSTEP; ++s) {
    const int buf = s & 1;
    wait_async0();          // this wave's tile-s copies have landed in LDS
    __syncthreads();        // everyone's landed; everyone done reading buf^1
    if (s + 1 < NSTEP) stage(s + 1, buf ^ 1);

    const int ph = s >> 4;

    // ---- A fragments (ISA 16-bit A 16x32 lane layout) ----
    v16bf afrag[2][2];
    #pragma unroll
    for (int mt = 0; mt < 2; ++mt) {
      const int row = wm * 32 + mt * 16 + (lane & 15);
      const int kb  = (lane >> 4) * 8;
      v8bf lo = *(const v8bf*)&lds_a1[buf][row * KB + kb];
      v8bf hi = *(const v8bf*)&lds_a1[buf][row * KB + 16 + kb];
      afrag[0][mt] = cat16(lo, hi);
      if (ph == 0) {
        v8bf lo2 = *(const v8bf*)&lds_a2[buf][row * KB + kb];
        v8bf hi2 = *(const v8bf*)&lds_a2[buf][row * KB + 16 + kb];
        afrag[1][mt] = cat16(lo2, hi2);
      } else {
        afrag[1][mt] = afrag[0][mt];
      }
    }
    // ---- B fragments (ISA 16-bit B 32x16 lane layout; LDS is N-major) ----
    v16bf bfrag[4];
    #pragma unroll
    for (int g = 0; g < 4; ++g) {
      const int col = wn * 16 + (lane & 15);
      const int kb2 = (lane >> 4) * 8;
      v8bf lo = *(const v8bf*)&lds_b[buf][g][col * KB + kb2];
      v8bf hi = *(const v8bf*)&lds_b[buf][g][col * KB + 16 + kb2];
      bfrag[g] = cat16(lo, hi);
    }

    // ---- 8 WMMAs per K-step per wave ----
    #pragma unroll
    for (int mt = 0; mt < 2; ++mt) {
      acc[0][mt] = __builtin_amdgcn_wmma_f32_16x16x32_bf16(false, afrag[0][mt], false, bfrag[0], (short)0, acc[0][mt], false, false);
      acc[3][mt] = __builtin_amdgcn_wmma_f32_16x16x32_bf16(false, afrag[0][mt], false, bfrag[3], (short)0, acc[3][mt], false, false);
      acc[1][mt] = __builtin_amdgcn_wmma_f32_16x16x32_bf16(false, afrag[1][mt], false, bfrag[1], (short)0, acc[1][mt], false, false);
      acc[2][mt] = __builtin_amdgcn_wmma_f32_16x16x32_bf16(false, afrag[1][mt], false, bfrag[2], (short)0, acc[2][mt], false, false);
    }
  }

  // ---- epilogue: biases + gates + cell update straight from accumulators ----
  const int col  = n_global + wn * 16 + (lane & 15);
  const int bofs = blk * BOUT + (col & (BOUT - 1));
  const float biasz = bz_[bofs] + rbz_[bofs];
  const float biasi = bi_[bofs] + rbi_[bofs];
  const float biasf = bf_[bofs] + rbf_[bofs];
  const float biaso = bo_[bofs] + rbo_[bofs];

  #pragma unroll
  for (int mt = 0; mt < 2; ++mt) {
    #pragma unroll
    for (int r = 0; r < 8; ++r) {
      const int row = m_base + wm * 32 + mt * 16 + (lane >> 4) * 8 + r;
      const size_t o = (size_t)row * HH + col;
      float zv = tanhf(acc[0][mt][r] + biasz);
      float iv = sigmoidf_(acc[1][mt][r] + biasi);
      float fv = sigmoidf_(acc[2][mt][r] + biasf);
      float ov = sigmoidf_(acc[3][mt][r] + biaso);
      float cv = fv * c_prev[o] + iv * zv;
      h_out[o] = ov * tanhf(cv);
      c_out[o] = cv;
    }
  }
}

// =====================================================================
extern "C" void kernel_launch(void* const* d_in, const int* in_sizes, int n_in,
                              void* d_out, int out_size, void* d_ws, size_t ws_size,
                              hipStream_t stream) {
  const float* x      = (const float*)d_in[0];
  const float* h_prev = (const float*)d_in[1];
  const float* c_prev = (const float*)d_in[2];
  const float* ln_g   = (const float*)d_in[3];
  const float* ln_b   = (const float*)d_in[4];
  const float* conv_w = (const float*)d_in[5];
  const float* conv_b = (const float*)d_in[6];
  const float* Wz = (const float*)d_in[7];  const float* bz = (const float*)d_in[8];
  const float* Wi = (const float*)d_in[9];  const float* bi = (const float*)d_in[10];
  const float* Wf = (const float*)d_in[11]; const float* bf = (const float*)d_in[12];
  const float* Wo = (const float*)d_in[13]; const float* bo = (const float*)d_in[14];
  const float* Rz = (const float*)d_in[15]; const float* rbz = (const float*)d_in[16];
  const float* Ri = (const float*)d_in[17]; const float* rbi = (const float*)d_in[18];
  const float* Rf = (const float*)d_in[19]; const float* rbf = (const float*)d_in[20];
  const float* Ro = (const float*)d_in[21]; const float* rbo = (const float*)d_in[22];

  float* h_out = (float*)d_out;
  float* c_out = (float*)d_out + (size_t)BB * HH;

  bf16* act = (bf16*)d_ws;             // 3 * S_ACT elems (48 MB)
  bf16* wt  = act + 3 * S_ACT;         // 2 * W_HALF elems (16 MB)

  prep_kernel<<<BB, 256, 0, stream>>>(x, h_prev, ln_g, ln_b, conv_w, conv_b, act);
  wcvt_kernel<<<dim3(16, 16, 32), 256, 0, stream>>>(Wz, Wi, Wf, Wo, Rz, Ri, Rf, Ro, wt);

  dim3 grid(HH / NT, BB / MT);   // (32, 64)
  gemm_gates_kernel<<<grid, 256, 0, stream>>>(
      act, wt, bz, bi, bf, bo, rbz, rbi, rbf, rbo, c_prev, h_out, c_out);
}
